// SkipConnection_25340307047020
// MI455X (gfx1250) — matrix operand order repaired and checked by reading, not compiled, not run on hardware
//
#include <hip/hip_runtime.h>
#include <hip/hip_bf16.h>

typedef __attribute__((ext_vector_type(16))) _Float16 v16h;
typedef __attribute__((ext_vector_type(8)))  _Float16 v8h;
typedef __attribute__((ext_vector_type(8)))  float    v8f;

#define BB 8
#define NN 784
#define CC 768
#define HH 8
#define DD 96
#define MP 800                 // N padded to multiple of 32 for WMMA K-loops
#define NBLK_MIX 19208         // (B*N*784)/256
#define CNT_F 4917248.0f       // B*N*N
#define SCALE_QK 0.10206207261596575f   // 96^-0.5

__device__ __forceinline__ v8f wmma16(v16h a, v16h b, v8f c) {
  return __builtin_amdgcn_wmma_f32_16x16x32_f16(false, a, false, b, (short)0, c, false, false);
}

// ---------------------------------------------------------------------------
// Kernel 1: per-token 3x3 same-pad conv over (3,16,16); writes f16 heads.
// q,k -> (B,H,N,D) row-major (n rows, d contiguous); v -> transposed (B,H,D,MP)
// ---------------------------------------------------------------------------
__global__ void qkv_conv(const float* __restrict__ q, const float* __restrict__ k,
                         const float* __restrict__ v,
                         const float* __restrict__ Wq, const float* __restrict__ Wk,
                         const float* __restrict__ Wv,
                         _Float16* __restrict__ qh, _Float16* __restrict__ kh,
                         _Float16* __restrict__ vt) {
  int token = blockIdx.x;            // b*N + n
  int which = blockIdx.y;            // 0=q 1=k 2=v
  int b = token / NN, n = token % NN;
  const float* x; const float* W;
  if (which == 0)      { x = q; W = Wq; }
  else if (which == 1) { x = k; W = Wk; }
  else                 { x = v; W = Wv; }

  __shared__ float sx[768];
  __shared__ float sw[81];
  int tid = threadIdx.x;
  for (int c = tid; c < 768; c += 256) sx[c] = x[(size_t)token * CC + c];
  if (tid < 81) sw[tid] = W[tid];
  __syncthreads();

  for (int c = tid; c < 768; c += 256) {
    int ch = c >> 8, pix = c & 255, y = pix >> 4, xx = pix & 15;
    float acc = 0.f;
    #pragma unroll
    for (int ci = 0; ci < 3; ci++)
      #pragma unroll
      for (int ky = 0; ky < 3; ky++) {
        int yy = y + ky - 1;
        if (yy < 0 || yy > 15) continue;
        #pragma unroll
        for (int kx = 0; kx < 3; kx++) {
          int xc = xx + kx - 1;
          if (xc < 0 || xc > 15) continue;
          acc += sw[((ch * 3 + ci) * 3 + ky) * 3 + kx] * sx[ci * 256 + yy * 16 + xc];
        }
      }
    int h = c / DD, d = c % DD;
    _Float16 hv = (_Float16)acc;
    if (which == 0)      qh[((size_t)(b * HH + h) * NN + n) * DD + d] = hv;
    else if (which == 1) kh[((size_t)(b * HH + h) * NN + n) * DD + d] = hv;
    else                 vt[((size_t)(b * HH + h) * DD + d) * MP + n] = hv;
  }
}

// Zero the pad columns (m = 784..799) of transposed V
__global__ void vt_pad(_Float16* __restrict__ vt) {
  int i = blockIdx.x * 256 + threadIdx.x;   // B*H*D*16 = 98304
  if (i < BB * HH * DD * 16) {
    int r = i >> 4, j = i & 15;
    vt[(size_t)r * MP + 784 + j] = (_Float16)0.f;
  }
}

// vsum[b,h,d] = sum_m V  (from transposed V, contiguous rows)
__global__ void vsum_k(const _Float16* __restrict__ vt, float* __restrict__ vsum) {
  int i = blockIdx.x * 256 + threadIdx.x;   // B*H*D = 6144
  if (i < BB * HH * DD) {
    const _Float16* row = vt + (size_t)i * MP;
    float s = 0.f;
    for (int m = 0; m < NN; m++) s += (float)row[m];
    vsum[i] = s;
  }
}

// fp32 -> f16 copy of Wp
__global__ void cvt_wp(const float* __restrict__ Wp, _Float16* __restrict__ wph) {
  int i = blockIdx.x * 256 + threadIdx.x;
  if (i < CC * CC) wph[i] = (_Float16)Wp[i];
}

// ---------------------------------------------------------------------------
// Kernel 2: S = Q K^T * scale (WMMA), row softmax over 784, write f16 probs
// grid (49, B*H), block 256 (8 waves). Each block: 16 rows x 784 cols.
// ---------------------------------------------------------------------------
__global__ void attn_softmax(const _Float16* __restrict__ qh,
                             const _Float16* __restrict__ kh,
                             _Float16* __restrict__ att) {
  int bh = blockIdx.y;
  int r0 = blockIdx.x * 16;
  int tid = threadIdx.x;
  int w = tid >> 5, l = tid & 31, lm = l & 15, hb = l >> 4;

  __shared__ float sc[16][800];   // 51.2 KB (784 used)

  // Preload A operands for kb = 0,32,64 (Q rows r0..r0+15, D=96)
  const _Float16* qrow = qh + ((size_t)bh * NN + r0 + lm) * DD + 16 * hb;
  v16h a0, a1, a2;
  *(v8h*)&a0       = *(const v8h*)(qrow);
  *((v8h*)&a0 + 1) = *(const v8h*)(qrow + 8);
  *(v8h*)&a1       = *(const v8h*)(qrow + 32);
  *((v8h*)&a1 + 1) = *(const v8h*)(qrow + 40);
  *(v8h*)&a2       = *(const v8h*)(qrow + 64);
  *((v8h*)&a2 + 1) = *(const v8h*)(qrow + 72);

  for (int ct = w; ct < 49; ct += 8) {
    int m = ct * 16 + lm;
    const _Float16* krow = kh + ((size_t)bh * NN + m) * DD + 16 * hb;
    v16h b0 = *(const v16h*)(krow);
    v16h b1 = *(const v16h*)(krow + 32);
    v16h b2 = *(const v16h*)(krow + 64);
    v8f c = {};
    c = wmma16(a0, b0, c);
    c = wmma16(a1, b1, c);
    c = wmma16(a2, b2, c);
    #pragma unroll
    for (int i = 0; i < 8; i++) sc[i + 8 * hb][ct * 16 + lm] = c[i] * SCALE_QK;
  }
  __syncthreads();

  // softmax: wave w handles rows 2w, 2w+1
  for (int rr = 0; rr < 2; rr++) {
    int row = w * 2 + rr;
    float mx = -1e30f;
    for (int cc = l; cc < NN; cc += 32) mx = fmaxf(mx, sc[row][cc]);
    #pragma unroll
    for (int off = 16; off; off >>= 1) mx = fmaxf(mx, __shfl_xor(mx, off, 32));
    float sum = 0.f;
    for (int cc = l; cc < NN; cc += 32) sum += __expf(sc[row][cc] - mx);
    #pragma unroll
    for (int off = 16; off; off >>= 1) sum += __shfl_xor(sum, off, 32);
    float inv = 1.f / sum;
    _Float16* orow = att + ((size_t)bh * NN + r0 + row) * MP;
    for (int cc = l; cc < NN; cc += 32)
      orow[cc] = (_Float16)(__expf(sc[row][cc] - mx) * inv);
    if (l < 16) orow[784 + l] = (_Float16)0.f;   // zero pad columns
  }
}

// ---------------------------------------------------------------------------
// Kernel 3: in-place cross-head mix A'[o] = sum_h Wre[o,h]*P[h] + bre[o],
// plus deterministic per-block partial sum / sumsq for BN stats.
// grid NBLK_MIX, block 256; one thread per (b,n,m) site, m < 784.
// ---------------------------------------------------------------------------
__global__ void mix_heads(_Float16* __restrict__ att,
                          const float* __restrict__ Wre, const float* __restrict__ bre,
                          float* __restrict__ psum, float* __restrict__ psq) {
  __shared__ float sW[64];
  __shared__ float sB[8];
  __shared__ float red[8][16];
  int tid = threadIdx.x;
  if (tid < 64) sW[tid] = Wre[tid];
  if (tid < 8)  sB[tid] = bre[tid];
  __syncthreads();

  int site = blockIdx.x * 256 + tid;          // exactly B*N*784 sites
  int b = site / (NN * 784);
  int rem = site % (NN * 784);
  int n = rem / 784, m = rem % 784;
  size_t base = ((size_t)(b * HH) * NN + n) * MP + m;
  const size_t hstride = (size_t)NN * MP;

  float p[8];
  #pragma unroll
  for (int h = 0; h < 8; h++) p[h] = (float)att[base + (size_t)h * hstride];

  float s[8], q2[8];
  #pragma unroll
  for (int o = 0; o < 8; o++) {
    float a = sB[o];
    #pragma unroll
    for (int h = 0; h < 8; h++) a += sW[o * 8 + h] * p[h];
    att[base + (size_t)o * hstride] = (_Float16)a;
    s[o] = a; q2[o] = a * a;
  }

  int l = tid & 31, w = tid >> 5;
  #pragma unroll
  for (int o = 0; o < 8; o++) {
    float ss = s[o], qq = q2[o];
    #pragma unroll
    for (int off = 16; off; off >>= 1) {
      ss += __shfl_xor(ss, off, 32);
      qq += __shfl_xor(qq, off, 32);
    }
    if (l == 0) { red[w][o] = ss; red[w][o + 8] = qq; }
  }
  __syncthreads();
  if (tid < 16) {
    float acc = 0.f;
    #pragma unroll
    for (int w2 = 0; w2 < 8; w2++) acc += red[w2][tid];
    if (tid < 8) psum[blockIdx.x * 8 + tid] = acc;
    else         psq[blockIdx.x * 8 + (tid - 8)] = acc;
  }
}

// Kernel 4: fixed-order final reduce -> alpha[o], const[o]
__global__ void bn_stats(const float* __restrict__ psum, const float* __restrict__ psq,
                         const float* __restrict__ gamma, const float* __restrict__ beta,
                         float* __restrict__ stats) {
  int tid = threadIdx.x;
  int o = tid >> 5, l = tid & 31;   // wave o handles head o
  float s = 0.f, q = 0.f;
  for (int i = l; i < NBLK_MIX; i += 32) {
    s += psum[i * 8 + o];
    q += psq[i * 8 + o];
  }
  #pragma unroll
  for (int off = 16; off; off >>= 1) {
    s += __shfl_xor(s, off, 32);
    q += __shfl_xor(q, off, 32);
  }
  if (l == 0) {
    float mean = s / CNT_F;
    float var  = q / CNT_F - mean * mean;
    float alpha = gamma[o] * rsqrtf(var + 1e-5f);
    stats[2 * o]     = alpha;
    stats[2 * o + 1] = beta[o] - alpha * mean;
  }
}

// ---------------------------------------------------------------------------
// Kernel 5: x = alpha*(A' @ V) + const*vsum  (WMMA over m, BN folded epilogue)
// grid (49, B*H), block 192 (6 waves, one 16-wide d-tile each)
// ---------------------------------------------------------------------------
__global__ void attn_v(const _Float16* __restrict__ att, const _Float16* __restrict__ vt,
                       const float* __restrict__ stats, const float* __restrict__ vsum,
                       _Float16* __restrict__ xh) {
  int bh = blockIdx.y;
  int b = bh >> 3, h = bh & 7;
  int r0 = blockIdx.x * 16;
  int tid = threadIdx.x;
  int w = tid >> 5, l = tid & 31, lm = l & 15, hb = l >> 4;
  int d0 = w * 16;

  const _Float16* arow = att + ((size_t)bh * NN + r0 + lm) * MP + 16 * hb;
  const _Float16* brow = vt + ((size_t)bh * DD + d0 + lm) * MP + 16 * hb;
  v8f c = {};
  for (int kb = 0; kb < MP; kb += 32) {
    v16h a;
    *(v8h*)&a       = *(const v8h*)(arow + kb);
    *((v8h*)&a + 1) = *(const v8h*)(arow + kb + 8);
    v16h bm = *(const v16h*)(brow + kb);
    c = wmma16(a, bm, c);
  }
  float alpha = stats[2 * h], cst = stats[2 * h + 1];
  float vs = vsum[(size_t)bh * DD + d0 + lm];
  #pragma unroll
  for (int i = 0; i < 8; i++) {
    int nrow = r0 + i + 8 * hb;
    float val = alpha * c[i] + cst * vs;
    xh[((size_t)b * NN + nrow) * CC + h * DD + d0 + lm] = (_Float16)val;
  }
}

// ---------------------------------------------------------------------------
// Kernel 6: out = xh @ Wp^T + bp  (fp32 output)
// grid (392, 6), block 256 (8 waves x 16-col tiles -> 128 cols per block)
// ---------------------------------------------------------------------------
__global__ void proj(const _Float16* __restrict__ xh, const _Float16* __restrict__ wph,
                     const float* __restrict__ bp, float* __restrict__ out) {
  int tid = threadIdx.x;
  int w = tid >> 5, l = tid & 31, lm = l & 15, hb = l >> 4;
  int r0 = blockIdx.x * 16;
  int o0 = blockIdx.y * 128 + w * 16;

  const _Float16* arow = xh + (size_t)(r0 + lm) * CC + 16 * hb;
  const _Float16* brow = wph + (size_t)(o0 + lm) * CC + 16 * hb;
  v8f c = {};
  for (int kb = 0; kb < CC; kb += 32) {
    v16h a;
    *(v8h*)&a       = *(const v8h*)(arow + kb);
    *((v8h*)&a + 1) = *(const v8h*)(arow + kb + 8);
    v16h bm = *(const v16h*)(brow + kb);
    c = wmma16(a, bm, c);
  }
  float bias = bp[o0 + lm];
  #pragma unroll
  for (int i = 0; i < 8; i++) {
    int row = r0 + i + 8 * hb;
    out[(size_t)row * CC + o0 + lm] = c[i] + bias;
  }
}

// ---------------------------------------------------------------------------
extern "C" void kernel_launch(void* const* d_in, const int* in_sizes, int n_in,
                              void* d_out, int out_size, void* d_ws, size_t ws_size,
                              hipStream_t stream) {
  (void)in_sizes; (void)n_in; (void)out_size; (void)ws_size;
  const float* q    = (const float*)d_in[0];
  const float* k    = (const float*)d_in[1];
  const float* v    = (const float*)d_in[2];
  const float* Wq   = (const float*)d_in[3];
  const float* Wk   = (const float*)d_in[4];
  const float* Wv   = (const float*)d_in[5];
  const float* Wre  = (const float*)d_in[6];
  const float* bre  = (const float*)d_in[7];
  const float* gam  = (const float*)d_in[8];
  const float* bet  = (const float*)d_in[9];
  const float* Wp   = (const float*)d_in[10];
  const float* bp   = (const float*)d_in[11];
  float* out = (float*)d_out;

  char* ws = (char*)d_ws;
  const size_t QKH  = (size_t)BB * HH * NN * DD * 2;      //  9,633,792
  const size_t VTSZ = (size_t)BB * HH * DD * MP * 2;      //  9,830,400
  const size_t ATTS = (size_t)BB * HH * NN * MP * 2;      // 80,281,600
  size_t off = 0;
  _Float16* qh   = (_Float16*)(ws + off); off += QKH;
  _Float16* kh   = (_Float16*)(ws + off); off += QKH;
  _Float16* vt   = (_Float16*)(ws + off); off += VTSZ;
  _Float16* att  = (_Float16*)(ws + off); off += ATTS;
  _Float16* xh   = (_Float16*)(ws + off); off += QKH;
  _Float16* wph  = (_Float16*)(ws + off); off += (size_t)CC * CC * 2;
  float* vsum    = (float*)(ws + off);    off += (size_t)BB * HH * DD * 4;
  float* psum    = (float*)(ws + off);    off += (size_t)NBLK_MIX * 8 * 4;
  float* psq     = (float*)(ws + off);    off += (size_t)NBLK_MIX * 8 * 4;
  float* stats   = (float*)(ws + off);    off += 64;

  qkv_conv    <<<dim3(BB * NN, 3), 256, 0, stream>>>(q, k, v, Wq, Wk, Wv, qh, kh, vt);
  vt_pad      <<<dim3(384), 256, 0, stream>>>(vt);
  vsum_k      <<<dim3(24), 256, 0, stream>>>(vt, vsum);
  cvt_wp      <<<dim3((CC * CC + 255) / 256), 256, 0, stream>>>(Wp, wph);
  attn_softmax<<<dim3(49, BB * HH), 256, 0, stream>>>(qh, kh, att);
  mix_heads   <<<dim3(NBLK_MIX), 256, 0, stream>>>(att, Wre, bre, psum, psq);
  bn_stats    <<<dim3(1), 256, 0, stream>>>(psum, psq, gam, bet, stats);
  attn_v      <<<dim3(49, BB * HH), 192, 0, stream>>>(att, vt, stats, vsum, xh);
  proj        <<<dim3(392, 6), 256, 0, stream>>>(xh, wph, bp, out);
}
